// DeiTSelfAttentionDropTokens_59579786330761
// MI455X (gfx1250) — compile-verified, matrix operand order checked by reading, not run on hardware
//
#include <hip/hip_runtime.h>
#include <hip/hip_bf16.h>
#include <math.h>

// ---------------- types / WMMA helpers (gfx1250, wave32) ----------------
typedef __attribute__((ext_vector_type(8)))  float  v8f;
typedef __attribute__((ext_vector_type(4)))  __bf16 v4bf;
typedef __attribute__((ext_vector_type(8)))  __bf16 v8bf;
typedef __attribute__((ext_vector_type(16))) __bf16 v16bf;

static __device__ __forceinline__ __bf16 f2bf(float f) { return (__bf16)f; }

static __device__ __forceinline__ v8f wmma_bf16(v16bf a, v16bf b, v8f c) {
  return __builtin_amdgcn_wmma_f32_16x16x32_bf16(false, a, false, b, (short)0, c, false, false);
}

// One 16x32-K fragment slice per lane: 8 halves at base+off0, 8 at base+off0+16
// (ISA 16-bit A/B layout: lanes<16 -> K 0..7 & 16..23; lanes>=16 -> 8..15 & 24..31).
static __device__ __forceinline__ v16bf load_frag(const __bf16* base, int off0) {
  union { v16bf v; v8bf h[2]; } u;
  u.h[0] = *(const v8bf*)(base + off0);
  u.h[1] = *(const v8bf*)(base + off0 + 16);
  return u.v;
}

// Wave-local LDS ordering (per-wave buffers; no block barrier).
static __device__ __forceinline__ void wave_lds_fence() {
  asm volatile("s_wait_dscnt 0" ::: "memory");
  __builtin_amdgcn_wave_barrier();
}

// Async global->LDS 16-byte copy (GLOBAL_LOAD_ASYNC_TO_LDS_B128, ASYNCcnt).
static __device__ __forceinline__ uint32_t ldsoff(const void* p) {
  // generic LDS address = {aperture_hi32, lds_offset_lo32}; low 32 bits = DS offset
  return (uint32_t)(uintptr_t)p;
}
static __device__ __forceinline__ void async_g2l_b128(uint32_t lds, const void* gptr) {
  asm volatile("global_load_async_to_lds_b128 %0, %1, off"
               :: "v"(lds), "v"(gptr) : "memory");
}
static __device__ __forceinline__ void wait_async() {
  asm volatile("s_wait_asynccnt 0" ::: "memory");
}

// ---------------- problem constants ----------------
#define BATCH 64
#define SEQ   198
#define CDIM  768
#define NH    12
#define HD    64
#define SM1   197          // S-1
#define KEEP  138          // ceil(0.7*197)
#define NKEEP 59           // 197-138
#define QT    13           // ceil(198/16) query/key tiles
#define KPAD  224          // 7*32 padded key dim for PV WMMA loop
#define MROWS (BATCH * SEQ)        // 12672
#define NX    (MROWS * CDIM)       // 9,732,096 elements

// ---------------- kernel: f32 -> bf16 flat convert (x4 vectorized) -------
__global__ void conv_x_kernel(const float* __restrict__ X, __bf16* __restrict__ Xb, int n4) {
  int i = blockIdx.x * blockDim.x + threadIdx.x;
  if (i >= n4) return;
  float4 x = ((const float4*)X)[i];
  v4bf o;
  o[0] = f2bf(x.x); o[1] = f2bf(x.y); o[2] = f2bf(x.z); o[3] = f2bf(x.w);
  ((v4bf*)Xb)[i] = o;
}

// ---------------- kernel: W -> Wt[which][n][k] bf16 (transpose) ----------
__global__ void conv_wt_kernel(const float* __restrict__ Wq, const float* __restrict__ Wk,
                               const float* __restrict__ Wv, __bf16* __restrict__ Wt) {
  int idx = blockIdx.x * blockDim.x + threadIdx.x;
  if (idx >= 3 * CDIM * CDIM) return;
  int which = idx / (CDIM * CDIM);
  int rem   = idx - which * (CDIM * CDIM);
  int n = rem / CDIM, k = rem - n * CDIM;     // writes coalesced; reads hit L2
  const float* W = (which == 0) ? Wq : (which == 1) ? Wk : Wv;
  Wt[idx] = f2bf(W[(size_t)k * CDIM + n]);
}

// ---------------- kernel: zero cls accumulator ----------------
__global__ void zero_cls_kernel(float* cls_ws, int n) {
  int i = blockIdx.x * blockDim.x + threadIdx.x;
  if (i < n) cls_ws[i] = 0.0f;
}

// ---------------- kernel: QKV projection GEMM (bf16 in, bf16 out) --------
// grid = (198 Mtiles, 12 Ntiles, 3 matrices), block = 256 (8 waves).
// 64x64 tile/block; wave w -> rows [mw*16,+16), cols [nw*32,+32).
// A/B tiles staged via async global->LDS b128; output [B][H][S][D] bf16.
__global__ void qkv_gemm_kernel(const __bf16* __restrict__ Xb, const __bf16* __restrict__ Wt,
                                const float* __restrict__ bq, const float* __restrict__ bk,
                                const float* __restrict__ bv,
                                __bf16* __restrict__ Qw, __bf16* __restrict__ Kw,
                                __bf16* __restrict__ Vw) {
  __shared__ __align__(16) __bf16 At[64 * 40];  // [64 rows][32 k], stride 40
  __shared__ __align__(16) __bf16 Bt[64 * 40];  // [64 cols][32 k]

  const int tid  = threadIdx.x;
  const int lane = tid & 31;
  const int w    = tid >> 5;
  const int mw   = w & 3;
  const int nw   = w >> 2;
  const int m0   = blockIdx.x * 64;
  const int n0   = blockIdx.y * 64;
  const int which = blockIdx.z;

  const __bf16* Wm = Wt + (size_t)which * CDIM * CDIM;
  const float*  Bm = (which == 0) ? bq : (which == 1) ? bk : bv;
  __bf16*       Om = (which == 0) ? Qw : (which == 1) ? Kw : Vw;

  // per-thread staging slots: 256 threads x one 16B chunk per tile
  const int sr = tid >> 2, sc8 = (tid & 3) * 8;          // row 0..63, half-chunk
  const uint32_t aLds = ldsoff(At + sr * 40 + sc8);
  const uint32_t bLds = ldsoff(Bt + sr * 40 + sc8);
  const __bf16* aG = Xb + (size_t)(m0 + sr) * CDIM + sc8;
  const __bf16* bG = Wm + (size_t)(n0 + sr) * CDIM + sc8;

  const int hi = (lane >= 16) ? 8 : 0;
  const __bf16* aP  = At + (mw * 16 + (lane & 15)) * 40;
  const __bf16* b0P = Bt + (nw * 32 + (lane & 15)) * 40;
  const __bf16* b1P = b0P + 16 * 40;

  v8f acc0 = {}, acc1 = {};
  for (int k0 = 0; k0 < CDIM; k0 += 32) {
    async_g2l_b128(aLds, aG + k0);
    async_g2l_b128(bLds, bG + k0);
    wait_async();
    __syncthreads();

    v16bf a  = load_frag(aP,  hi);
    v16bf b0 = load_frag(b0P, hi);
    v16bf b1 = load_frag(b1P, hi);
    acc0 = wmma_bf16(a, b0, acc0);
    acc1 = wmma_bf16(a, b1, acc1);

    wave_lds_fence();      // frag ds_loads done before next async overwrite
    __syncthreads();
  }

  // epilogue: bias + bf16 scatter to [B][H][S][D]
  const int rhi = (lane >> 4) << 3;
  for (int t = 0; t < 2; ++t) {
    int col = n0 + nw * 32 + t * 16 + (lane & 15);
    int h = col >> 6, d = col & 63;
    float bias = Bm[col];
    v8f acc = t ? acc1 : acc0;
    for (int r = 0; r < 8; ++r) {
      int m = m0 + mw * 16 + r + rhi;          // always < 12672
      int bb = m / SEQ, s = m - bb * SEQ;
      Om[(((size_t)bb * NH + h) * SEQ + s) * HD + d] = f2bf(acc[r] + bias);
    }
  }
}

// ---------------- kernel: attention per (bh, qtile) ----------------
// block = 128 (4 waves). Phase A: Q*K^T tiles via WMMA -> LDS f32 scores.
// f32 softmax; cls row accumulated across heads with atomics. Phase B: P*V.
__global__ void attn_kernel(const __bf16* __restrict__ Qw,
                            const __bf16* __restrict__ Kw,
                            const __bf16* __restrict__ Vw,
                            float* __restrict__ ctx_out,
                            float* __restrict__ cls_ws) {
  __shared__ __align__(16) __bf16 Qs[16 * 72];        // q tile
  __shared__ __align__(16) __bf16 Ks[4 * 16 * 72];    // per-wave key tile
  __shared__ __align__(16) __bf16 Vt[4 * 16 * 40];    // per-wave V^T tile
  __shared__ __align__(16) __bf16 Pb[16 * 232];       // probs bf16
  __shared__ float Sc[16 * 224];                      // scores/probs f32

  const int tid  = threadIdx.x;
  const int lane = tid & 31;
  const int w    = tid >> 5;
  const int bh   = blockIdx.y;
  const int bb   = bh / NH;
  const int hh   = bh - bb * NH;
  const int q0   = blockIdx.x * 16;
  const int hi   = (lane >= 16) ? 8 : 0;
  const int rhi  = (lane >> 4) << 3;

  const __bf16* Qh = Qw + (size_t)bh * SEQ * HD;
  const __bf16* Kh = Kw + (size_t)bh * SEQ * HD;
  const __bf16* Vh = Vw + (size_t)bh * SEQ * HD;

  // stage Q tile: 128 chunks of 16B, one per thread; invalid rows zeroed
  {
    int r = tid >> 3, c8 = (tid & 7) * 8, s = q0 + r;
    __bf16* dst = Qs + r * 72 + c8;
    if (s < SEQ) {
      async_g2l_b128(ldsoff(dst), Qh + (size_t)s * HD + c8);
    } else {
      v8bf z; for (int q = 0; q < 8; ++q) z[q] = f2bf(0.0f);
      *(v8bf*)dst = z;
    }
  }
  wait_async();
  __syncthreads();

  // Phase A: waves strided over 13 key tiles (tail keys>=SEQ never read later)
  __bf16* Ksw = Ks + w * 16 * 72;
  for (int j = w; j < QT; j += 4) {
    for (int i = 0; i < 4; ++i) {
      int e = i * 32 + lane, kk = e >> 3, c8 = (e & 7) * 8;
      async_g2l_b128(ldsoff(Ksw + kk * 72 + c8),
                     Kh + (size_t)(j * 16 + kk) * HD + c8);
    }
    wait_async();
    v8f acc = {};
    for (int d0 = 0; d0 < HD; d0 += 32) {
      v16bf a = load_frag(Qs  + (lane & 15) * 72 + d0, hi);
      v16bf b = load_frag(Ksw + (lane & 15) * 72 + d0, hi);
      acc = wmma_bf16(a, b, acc);
    }
    int col = j * 16 + (lane & 15);
    for (int r = 0; r < 8; ++r)
      Sc[(r + rhi) * 224 + col] = acc[r] * 0.125f;   // 1/sqrt(64)
    wave_lds_fence();                                 // before Ksw reuse
  }
  __syncthreads();

  // softmax: one thread per query row, f32, keys < SEQ only
  if (tid < 16) {
    float* row = Sc + tid * 224;
    float m = -INFINITY;
    for (int k = 0; k < SEQ; ++k) m = fmaxf(m, row[k]);
    float sum = 0.0f;
    for (int k = 0; k < SEQ; ++k) { float e = expf(row[k] - m); row[k] = e; sum += e; }
    float inv = 1.0f / sum;
    __bf16* pr = Pb + tid * 232;
    for (int k = 0; k < SEQ; ++k) { float p = row[k] * inv; row[k] = p; pr[k] = f2bf(p); }
    for (int k = SEQ; k < KPAD; ++k) pr[k] = f2bf(0.0f);
  }
  __syncthreads();

  // cls scores: probs[query 0][1:] averaged over heads
  if (blockIdx.x == 0) {
    for (int k = 1 + tid; k < SEQ; k += 128)
      atomicAdd(&cls_ws[(size_t)bb * SM1 + (k - 1)], Sc[k] * (1.0f / (float)NH));
  }

  // Phase B: ctx = P*V ; wave w owns d columns [w*16,+16)
  const int n0 = w * 16;
  __bf16* Vtw = Vt + w * 16 * 40;
  v8f acc2 = {};
  for (int c = 0; c < KPAD / 32; ++c) {
    int kk0 = c * 32;
    for (int i = 0; i < 2; ++i) {                 // 64 chunks / 32 lanes
      int e = i * 32 + lane, key = e >> 1, cc = (e & 1) * 8;
      int gk = kk0 + key;
      v8bf vv;
      if (gk < SEQ) vv = *(const v8bf*)(Vh + (size_t)gk * HD + n0 + cc);
      else          for (int q = 0; q < 8; ++q) vv[q] = f2bf(0.0f);
      for (int q = 0; q < 8; ++q) Vtw[(cc + q) * 40 + key] = vv[q];  // transpose
    }
    wave_lds_fence();
    v16bf a = load_frag(Pb  + (lane & 15) * 232 + kk0, hi);
    v16bf b = load_frag(Vtw + (lane & 15) * 40, hi);
    acc2 = wmma_bf16(a, b, acc2);
    wave_lds_fence();
  }
  for (int r = 0; r < 8; ++r) {
    int s = q0 + r + rhi;
    if (s < SEQ)
      ctx_out[((size_t)bb * SEQ + s) * CDIM + hh * HD + n0 + (lane & 15)] = acc2[r];
  }
}

// ---------------- kernel: cls copy + top-k + residual ----------------
__global__ void topk_kernel(const float* __restrict__ cls_ws,
                            float* __restrict__ out_cls,
                            float* __restrict__ out_top,
                            float* __restrict__ out_non) {
  __shared__ float vals[SM1];
  __shared__ int   istop[SM1];
  const int b = blockIdx.x;
  const int t = threadIdx.x;

  if (t < SM1) vals[t] = cls_ws[(size_t)b * SM1 + t];
  __syncthreads();

  if (t < SM1) {
    float v = vals[t];
    int rank = 0;
    for (int j = 0; j < SM1; ++j) {
      float u = vals[j];
      rank += (u > v) || (u == v && j < t);      // stable descending = lax.top_k
    }
    istop[t] = (rank < KEEP);
    if (rank < KEEP) out_top[(size_t)b * KEEP + rank] = (float)t;
    out_cls[(size_t)b * SM1 + t] = v;
  }
  __syncthreads();

  if (t < SM1 && !istop[t]) {
    int pos = 0;
    for (int j = 0; j < t; ++j) pos += !istop[j];
    out_non[(size_t)b * NKEEP + pos] = (float)t;  // ascending residual indices
  }
}

// ---------------- launcher ----------------
extern "C" void kernel_launch(void* const* d_in, const int* in_sizes, int n_in,
                              void* d_out, int out_size, void* d_ws, size_t ws_size,
                              hipStream_t stream) {
  (void)in_sizes; (void)n_in; (void)out_size; (void)ws_size;
  const float* X  = (const float*)d_in[0];
  const float* Wq = (const float*)d_in[1];
  const float* bq = (const float*)d_in[2];
  const float* Wk = (const float*)d_in[3];
  const float* bk = (const float*)d_in[4];
  const float* Wv = (const float*)d_in[5];
  const float* bv = (const float*)d_in[6];

  const size_t perMat = (size_t)BATCH * NH * SEQ * HD;   // 9,732,096
  __bf16* Xb = (__bf16*)d_ws;
  __bf16* Wt = Xb + NX;
  __bf16* Qw = Wt + (size_t)3 * CDIM * CDIM;
  __bf16* Kw = Qw + perMat;
  __bf16* Vw = Kw + perMat;
  float* cls_ws = (float*)(Vw + perMat);

  float* out   = (float*)d_out;
  float* o_ctx = out;                                    // B*S*C
  float* o_cls = o_ctx + (size_t)BATCH * SEQ * CDIM;     // B*197
  float* o_top = o_cls + (size_t)BATCH * SM1;            // B*138
  float* o_non = o_top + (size_t)BATCH * KEEP;           // B*59

  const int n4 = NX / 4;
  conv_x_kernel<<<(n4 + 255) / 256, 256, 0, stream>>>(X, Xb, n4);

  const int nw = 3 * CDIM * CDIM;
  conv_wt_kernel<<<(nw + 255) / 256, 256, 0, stream>>>(Wq, Wk, Wv, Wt);

  const int nCls = BATCH * SM1;
  zero_cls_kernel<<<(nCls + 255) / 256, 256, 0, stream>>>(cls_ws, nCls);

  qkv_gemm_kernel<<<dim3(SEQ, CDIM / 64, 3), 256, 0, stream>>>(
      Xb, Wt, bq, bk, bv, Qw, Kw, Vw);

  attn_kernel<<<dim3(QT, BATCH * NH), 128, 0, stream>>>(Qw, Kw, Vw, o_ctx, cls_ws);

  topk_kernel<<<BATCH, 256, 0, stream>>>(cls_ws, o_cls, o_top, o_non);
}
